// Classifier_71141838291631
// MI455X (gfx1250) — compile-verified, matrix operand order checked by reading, not run on hardware
//
#include <hip/hip_runtime.h>
#include <hip/hip_bf16.h>

typedef __attribute__((ext_vector_type(16))) _Float16 v16h;
typedef __attribute__((ext_vector_type(8)))  _Float16 v8h;
typedef __attribute__((ext_vector_type(4)))  _Float16 v4h;
typedef __attribute__((ext_vector_type(8)))  float    v8f;

#define N_ROIS   1000
#define DEPTH    256
#define RANGE_C  64
#define CIN      (DEPTH + RANGE_C)      // 320
#define POOL     7
#define K1       (CIN * POOL * POOL)    // 15680
#define HID      1024
#define BN_EPS   1e-3f

// ---------------------------------------------------------------------------
// f32 -> f16 conversion (vectorized by 4)
// ---------------------------------------------------------------------------
__global__ __launch_bounds__(256) void cvt_f32_to_f16(const float* __restrict__ src,
                                                      _Float16* __restrict__ dst, int n4) {
  int i = blockIdx.x * blockDim.x + threadIdx.x;
  if (i >= n4) return;
  float4 v = ((const float4*)src)[i];
  v4h h = { (_Float16)v.x, (_Float16)v.y, (_Float16)v.z, (_Float16)v.w };
  ((v4h*)dst)[i] = h;
}

// ---------------------------------------------------------------------------
// ROI-align (TF crop_and_resize, bilinear, extrapolation 0) with FPN level
// selection + range crop.  One block per ROI; writes f32 to d_out (output #5)
// and f16 to workspace (GEMM1 A matrix).
// ---------------------------------------------------------------------------
__global__ __launch_bounds__(256) void roi_align_kernel(
    const float* __restrict__ p2, const float* __restrict__ p3,
    const float* __restrict__ p4, const float* __restrict__ p5,
    const float* __restrict__ rois, const float* __restrict__ ranges,
    float* __restrict__ outF, _Float16* __restrict__ outH) {
  int n = blockIdx.x;
  float by1 = rois[n * 4 + 0], bx1 = rois[n * 4 + 1];
  float by2 = rois[n * 4 + 2], bx2 = rois[n * 4 + 3];
  float h = by2 - by1, w = bx2 - bx1;
  // roi_level = clip(round(4 + log2(sqrt(h*w)*640/224)), 2, 5)
  float rl = 4.0f + log2f(sqrtf(h * w) * (640.0f / 224.0f));
  int level = (int)rintf(rl);
  level = level < 2 ? 2 : (level > 5 ? 5 : level);
  const float* fm; int H;
  if (level == 2)      { fm = p2; H = 160; }
  else if (level == 3) { fm = p3; H = 80;  }
  else if (level == 4) { fm = p4; H = 40;  }
  else                 { fm = p5; H = 20;  }

  for (int idx = threadIdx.x; idx < K1; idx += 256) {
    int c   = idx / 49;
    int rem = idx - c * 49;
    int py  = rem / 7;
    int px  = rem - py * 7;
    const float* F; int HH; int ch;
    if (c < DEPTH) { F = fm;     HH = H;   ch = c; }
    else           { F = ranges; HH = 160; ch = c - DEPTH; }
    float Hm1 = (float)(HH - 1);
    float in_y = by1 * Hm1 + (float)py * (h * Hm1 * (1.0f / 6.0f));
    float in_x = bx1 * Hm1 + (float)px * (w * Hm1 * (1.0f / 6.0f));
    float top  = floorf(in_y), bot   = ceilf(in_y);
    float left = floorf(in_x), right = ceilf(in_x);
    float ly = in_y - top, lx = in_x - left;
    int ti = (int)fminf(fmaxf(top,   0.f), Hm1);
    int bi = (int)fminf(fmaxf(bot,   0.f), Hm1);
    int li = (int)fminf(fmaxf(left,  0.f), Hm1);
    int ri = (int)fminf(fmaxf(right, 0.f), Hm1);
    const float* Fc = F + (size_t)ch * HH * HH;
    float tl = Fc[ti * HH + li], tr = Fc[ti * HH + ri];
    float bl = Fc[bi * HH + li], br = Fc[bi * HH + ri];
    float top_v = tl + (tr - tl) * lx;
    float bot_v = bl + (br - bl) * lx;
    float v = top_v + (bot_v - top_v) * ly;
    bool valid = (in_y >= 0.f) && (in_y <= Hm1) && (in_x >= 0.f) && (in_x <= Hm1);
    v = valid ? v : 0.f;
    size_t o = (size_t)n * K1 + idx;
    outF[o] = v;
    outH[o] = (_Float16)v;
  }
}

// ---------------------------------------------------------------------------
// WMMA f16 GEMM:  C[M,N] = A[M,K] * B[N,K]^T, fused bias + BN + ReLU.
// Register-blocked: one wave -> 32x64 output tile (2x4 wmma tiles, 8 x v8f
// accumulators).  Per K-step of 32: 2 A frags + 4 B frags (12 b128 loads)
// feed 8 v_wmma_f32_16x16x32_f16 -> 1.5 loads per wmma.
// Per-lane fragment loads follow the ISA 16-bit A/B layout:
//   lanes 0-15 : K = k..k+7  and k+16..k+23   (two b128 loads)
//   lanes 16-31: K = k+8..k+15 and k+24..k+31
// Block = 4 waves (2Mx2N) -> 64(M) x 128(N) tile. Grid = (ceil(M/64), N/128).
// ---------------------------------------------------------------------------
union V16HU { v16h v; v8h h[2]; };

__global__ __launch_bounds__(128) void wmma_gemm_bn_relu(
    const _Float16* __restrict__ A,   // [M, K] (pad rows clamped)
    const _Float16* __restrict__ B,   // [N, K]
    const float* __restrict__ bias,
    const float* __restrict__ bn_gamma, const float* __restrict__ bn_beta,
    const float* __restrict__ bn_mean,  const float* __restrict__ bn_var,
    float* __restrict__ outF, _Float16* __restrict__ outH,
    int M, int N, int K, int storeHalf) {
  int wave = threadIdx.x >> 5;
  int lane = threadIdx.x & 31;
  int hi   = lane >> 4;          // half-wave select (K-group)
  int lr   = lane & 15;
  int wm   = wave >> 1;          // 2x2 wave arrangement in block
  int wn   = wave & 1;
  int m0 = blockIdx.x * 64 + wm * 32;
  int n0 = blockIdx.y * 128 + wn * 64;

  int r0 = m0 + lr;      if (r0 > M - 1) r0 = M - 1;   // clamp pad rows
  int r1 = m0 + 16 + lr; if (r1 > M - 1) r1 = M - 1;
  const _Float16* ap0 = A + (size_t)r0 * K + hi * 8;
  const _Float16* ap1 = A + (size_t)r1 * K + hi * 8;
  const _Float16* bp0 = B + (size_t)(n0 + lr) * K + hi * 8;
  const _Float16* bp1 = B + (size_t)(n0 + 16 + lr) * K + hi * 8;
  const _Float16* bp2 = B + (size_t)(n0 + 32 + lr) * K + hi * 8;
  const _Float16* bp3 = B + (size_t)(n0 + 48 + lr) * K + hi * 8;

  v8f acc[2][4] = {};
  for (int k = 0; k < K; k += 32) {
    V16HU a[2], b[4];
    a[0].h[0] = *(const v8h*)(ap0 + k);
    a[0].h[1] = *(const v8h*)(ap0 + k + 16);
    b[0].h[0] = *(const v8h*)(bp0 + k);
    b[0].h[1] = *(const v8h*)(bp0 + k + 16);
    a[1].h[0] = *(const v8h*)(ap1 + k);
    a[1].h[1] = *(const v8h*)(ap1 + k + 16);
    b[1].h[0] = *(const v8h*)(bp1 + k);
    b[1].h[1] = *(const v8h*)(bp1 + k + 16);
    b[2].h[0] = *(const v8h*)(bp2 + k);
    b[2].h[1] = *(const v8h*)(bp2 + k + 16);
    b[3].h[0] = *(const v8h*)(bp3 + k);
    b[3].h[1] = *(const v8h*)(bp3 + k + 16);
#pragma unroll
    for (int j = 0; j < 4; ++j)
#pragma unroll
      for (int i = 0; i < 2; ++i)
        acc[i][j] = __builtin_amdgcn_wmma_f32_16x16x32_f16(
            false, a[i].v, false, b[j].v, (short)0, acc[i][j], false, false);
  }

#pragma unroll
  for (int j = 0; j < 4; ++j) {
    int col = n0 + j * 16 + lr;
    float bb    = bias[col];
    float mean  = bn_mean[col];
    float scale = bn_gamma[col] / sqrtf(bn_var[col] + BN_EPS);
    float beta  = bn_beta[col];
#pragma unroll
    for (int i = 0; i < 2; ++i) {
#pragma unroll
      for (int r = 0; r < 8; ++r) {
        int row = m0 + i * 16 + r + hi * 8;
        if (row < M) {
          float x = acc[i][j][r] + bb;
          x = (x - mean) * scale + beta;
          x = x > 0.f ? x : 0.f;
          if (storeHalf) outH[(size_t)row * N + col] = (_Float16)x;
          else           outF[(size_t)row * N + col] = x;
        }
      }
    }
  }
}

// ---------------------------------------------------------------------------
// Heads: 16 dot products of length 1024 per ROI (2 logits, 8 bbox, 6 params)
// ---------------------------------------------------------------------------
__global__ __launch_bounds__(256) void heads_kernel(
    const float* __restrict__ X,                       // [1000,1024]
    const float* __restrict__ wc, const float* __restrict__ bc,
    const float* __restrict__ wb, const float* __restrict__ bb,
    const float* __restrict__ wp, const float* __restrict__ bp,
    float* __restrict__ logits, float* __restrict__ bbox,
    float* __restrict__ params) {
  int t = blockIdx.x * blockDim.x + threadIdx.x;
  if (t >= N_ROIS * 16) return;
  int n = t >> 4, j = t & 15;
  const float* wr; float b; float* dst;
  if (j < 2)       { wr = wc + j * HID;        b = bc[j];      dst = logits + n * 2 + j; }
  else if (j < 10) { int q = j - 2;  wr = wb + q * HID; b = bb[q]; dst = bbox   + n * 8 + q; }
  else             { int q = j - 10; wr = wp + q * HID; b = bp[q]; dst = params + n * 6 + q; }
  const float* x = X + (size_t)n * HID;
  float s = 0.f;
#pragma unroll 4
  for (int k = 0; k < HID; ++k) s += x[k] * wr[k];
  *dst = s + b;
}

__global__ __launch_bounds__(256) void softmax2_kernel(const float* __restrict__ logits,
                                                       float* __restrict__ probs) {
  int n = blockIdx.x * blockDim.x + threadIdx.x;
  if (n >= N_ROIS) return;
  float a = logits[n * 2], b = logits[n * 2 + 1];
  float m = fmaxf(a, b);
  float ea = expf(a - m), eb = expf(b - m);
  float inv = 1.0f / (ea + eb);
  probs[n * 2]     = ea * inv;
  probs[n * 2 + 1] = eb * inv;
}

// ---------------------------------------------------------------------------
extern "C" void kernel_launch(void* const* d_in, const int* in_sizes, int n_in,
                              void* d_out, int out_size, void* d_ws, size_t ws_size,
                              hipStream_t stream) {
  const float* p2     = (const float*)d_in[0];
  const float* p3     = (const float*)d_in[1];
  const float* p4     = (const float*)d_in[2];
  const float* p5     = (const float*)d_in[3];
  const float* rois   = (const float*)d_in[4];
  const float* ranges = (const float*)d_in[5];
  const float* w1     = (const float*)d_in[6];
  const float* b1     = (const float*)d_in[7];
  const float* g1     = (const float*)d_in[8];
  const float* be1    = (const float*)d_in[9];
  const float* mu1    = (const float*)d_in[10];
  const float* var1   = (const float*)d_in[11];
  const float* w2     = (const float*)d_in[12];
  const float* b2     = (const float*)d_in[13];
  const float* g2     = (const float*)d_in[14];
  const float* be2    = (const float*)d_in[15];
  const float* mu2    = (const float*)d_in[16];
  const float* var2   = (const float*)d_in[17];
  const float* wc     = (const float*)d_in[18];
  const float* bc     = (const float*)d_in[19];
  const float* wb     = (const float*)d_in[20];
  const float* bb     = (const float*)d_in[21];
  const float* wp     = (const float*)d_in[22];
  const float* bp     = (const float*)d_in[23];

  // output layout (flat f32 concat): logits[2000] probs[2000] bbox[8000]
  // params[6000] roi_features[15,680,000]
  float* out      = (float*)d_out;
  float* o_logits = out;
  float* o_probs  = out + 2000;
  float* o_bbox   = out + 4000;
  float* o_params = out + 12000;
  float* o_roif   = out + 18000;

  // workspace layout (16B aligned)
  char* ws = (char*)d_ws;
  const size_t W1_ELEMS = (size_t)HID * K1;       // 16,056,320
  const size_t W2_ELEMS = (size_t)HID * HID;      //  1,048,576
  _Float16* W1h = (_Float16*)ws;                                   // 32,112,640 B
  _Float16* W2h = (_Float16*)(ws + 32112640);                      //  2,097,152 B
  _Float16* Ah  = (_Float16*)(ws + 34209792);                      // 31,360,000 B
  _Float16* X1h = (_Float16*)(ws + 65569792);                      //  2,048,000 B
  float*    X2  = (float*)   (ws + 67617792);                      //  4,096,000 B

  // 1-2) weight conversion to f16
  {
    int n4 = (int)(W1_ELEMS / 4);
    cvt_f32_to_f16<<<(n4 + 255) / 256, 256, 0, stream>>>(w1, W1h, n4);
    int m4 = (int)(W2_ELEMS / 4);
    cvt_f32_to_f16<<<(m4 + 255) / 256, 256, 0, stream>>>(w2, W2h, m4);
  }

  // 3) ROI align -> f32 output + f16 GEMM operand
  roi_align_kernel<<<N_ROIS, 256, 0, stream>>>(p2, p3, p4, p5, rois, ranges,
                                               o_roif, Ah);

  // 4) GEMM1: [1000,15680] x [15680,1024] + bias + BN + ReLU -> f16 X1h
  {
    dim3 grid((N_ROIS + 63) / 64, HID / 128);
    wmma_gemm_bn_relu<<<grid, 128, 0, stream>>>(Ah, W1h, b1, g1, be1, mu1, var1,
                                                nullptr, X1h, N_ROIS, HID, K1, 1);
  }

  // 5) GEMM2: [1000,1024] x [1024,1024] + bias + BN + ReLU -> f32 X2
  {
    dim3 grid((N_ROIS + 63) / 64, HID / 128);
    wmma_gemm_bn_relu<<<grid, 128, 0, stream>>>(X1h, W2h, b2, g2, be2, mu2, var2,
                                                X2, nullptr, N_ROIS, HID, HID, 0);
  }

  // 6) heads
  heads_kernel<<<(N_ROIS * 16 + 255) / 256, 256, 0, stream>>>(
      X2, wc, bc, wb, bb, wp, bp, o_logits, o_bbox, o_params);

  // 7) softmax over 2 classes
  softmax2_kernel<<<(N_ROIS + 255) / 256, 256, 0, stream>>>(o_logits, o_probs);

  (void)in_sizes; (void)n_in; (void)out_size; (void)ws_size;
}
